// DecoderLSTMAttention_5351529251414
// MI455X (gfx1250) — matrix operand order repaired and checked by reading, not compiled, not run on hardware
//
#include <hip/hip_runtime.h>
#include <stdint.h>

// ---------------------------------------------------------------------------
// Types for CDNA5 WMMA
// ---------------------------------------------------------------------------
typedef __attribute__((ext_vector_type(16))) __bf16 v16bf;
typedef __attribute__((ext_vector_type(8)))  float  v8f;

#define BM 64
#define BN 128
#define BK 64      // K elements staged per LDS buffer (2 WMMA k-steps)
#define LDSS 72    // padded LDS row stride in ushorts (64 data + 8 pad) = 144B (16B-aligned)

__device__ __forceinline__ unsigned short f32_to_bf16(float x) {
    union { float f; uint32_t u; } c; c.f = x;
    uint32_t u = c.u;
    return (unsigned short)((u + 0x7FFFu + ((u >> 16) & 1u)) >> 16);  // RNE
}

// gfx1250 async copy: global memory -> LDS, 16B per lane, tracked by ASYNCcnt.
__device__ __forceinline__ void async_ld_b128(unsigned lds_addr, const void* gptr) {
    asm volatile("global_load_async_to_lds_b128 %0, %1, off"
                 :: "v"(lds_addr), "v"((unsigned long long)(uintptr_t)gptr)
                 : "memory");
}

// Issue one 64x64 A-slab (2 chunks/thread) + 128x64 B-slab (4 chunks/thread).
__device__ __forceinline__ void issue_tile_async(
    const unsigned short* __restrict__ A, int lda,
    const unsigned short* __restrict__ B, int ldb,
    int blockM, int blockN, int k0, int tid,
    unsigned asBase, unsigned bsBase)
{
    #pragma unroll
    for (int i = 0; i < 2; ++i) {                 // A: 64 rows x 8 chunks = 512
        int id = tid + 256 * i;
        int r = id >> 3, s = id & 7;
        unsigned lds = asBase + (unsigned)(r * LDSS + s * 8) * 2u;
        async_ld_b128(lds, A + (size_t)(blockM + r) * lda + k0 + s * 8);
    }
    #pragma unroll
    for (int i = 0; i < 4; ++i) {                 // B: 128 rows x 8 chunks = 1024
        int id = tid + 256 * i;
        int r = id >> 3, s = id & 7;
        unsigned lds = bsBase + (unsigned)(r * LDSS + s * 8) * 2u;
        async_ld_b128(lds, B + (size_t)(blockN + r) * ldb + k0 + s * 8);
    }
}

// ---------------------------------------------------------------------------
// bf16 WMMA GEMM:  C[M,N] = A[M,K] * B[N,K]^T  (+ bias[n]) (+ addend[m,n])
// A,B row-major, K contiguous. M%64==0, N%128==0, K%64==0 required.
// Double-buffered LDS filled by global_load_async_to_lds_b128.
// ---------------------------------------------------------------------------
__global__ __launch_bounds__(256)
void wmma_gemm_bf16(const unsigned short* __restrict__ A, int lda,
                    const unsigned short* __restrict__ B, int ldb,
                    float* __restrict__ C, int ldc,
                    const float* __restrict__ bias,              // nullable, len N
                    const float* __restrict__ addend, int ldadd, // nullable
                    int K)
{
    __shared__ unsigned short As[2][BM * LDSS];
    __shared__ unsigned short Bs[2][BN * LDSS];

    const int tid  = threadIdx.x;
    const int lane = tid & 31;
    const int wave = tid >> 5;
    const int wm   = wave & 1;   // 2 waves along M
    const int wn   = wave >> 1;  // 4 waves along N
    const int blockM = blockIdx.y * BM;
    const int blockN = blockIdx.x * BN;

    const unsigned asBase = (unsigned)(uintptr_t)&As[0][0];
    const unsigned bsBase = (unsigned)(uintptr_t)&Bs[0][0];
    const unsigned ABUF = BM * LDSS * 2u;   // bytes per A stage
    const unsigned BBUF = BN * LDSS * 2u;   // bytes per B stage

    v8f acc[2][2] = {};

    const int nk = K / BK;
    issue_tile_async(A, lda, B, ldb, blockM, blockN, 0, tid, asBase, bsBase);

    for (int ki = 0; ki < nk; ++ki) {
        const int buf = ki & 1;
        if (ki + 1 < nk) {
            // prefetch next slab into the other stage, then wait for current stage
            issue_tile_async(A, lda, B, ldb, blockM, blockN, (ki + 1) * BK, tid,
                             asBase + (unsigned)(buf ^ 1) * ABUF,
                             bsBase + (unsigned)(buf ^ 1) * BBUF);
            if (ki + 2 < nk)   // pull the slab-after-next toward L2
                __builtin_prefetch(B + (size_t)(blockN + (tid >> 3)) * ldb + (ki + 2) * BK, 0, 1);
            asm volatile("s_wait_asynccnt 0x6" ::: "memory");
        } else {
            asm volatile("s_wait_asynccnt 0x0" ::: "memory");
        }
        __syncthreads();   // current stage visible to all waves

        const unsigned short* as = &As[buf][0];
        const unsigned short* bs = &Bs[buf][0];

        #pragma unroll
        for (int kk = 0; kk < 2; ++kk) {          // two 16x16x32 k-steps per slab
            const int kb = kk * 32;
            union F { uint4 q[2]; v16bf v; } aF[2], bF[2];
            #pragma unroll
            for (int i = 0; i < 2; ++i) {
                // A 16x32 frag: lanes 0-15 hold K0-7 & K16-23; lanes 16-31 hold K8-15 & K24-31
                int m  = wm * 32 + i * 16 + (lane & 15);
                int ko = kb + ((lane < 16) ? 0 : 8);
                aF[i].q[0] = *(const uint4*)(as + m * LDSS + ko);
                aF[i].q[1] = *(const uint4*)(as + m * LDSS + ko + 16);
            }
            #pragma unroll
            for (int j = 0; j < 2; ++j) {
                // B 32x16 frag: lanes 0-15 hold K0-7 & K16-23 of col n; lanes 16-31 hold K8-15 & K24-31
                int n  = wn * 32 + j * 16 + (lane & 15);
                int ko = kb + ((lane < 16) ? 0 : 8);
                bF[j].q[0] = *(const uint4*)(bs + n * LDSS + ko);
                bF[j].q[1] = *(const uint4*)(bs + n * LDSS + ko + 16);
            }
            #pragma unroll
            for (int i = 0; i < 2; ++i)
                #pragma unroll
                for (int j = 0; j < 2; ++j)
                    acc[i][j] = __builtin_amdgcn_wmma_f32_16x16x32_bf16(
                        false, aF[i].v, false, bF[j].v, (short)0, acc[i][j], false, false);
        }
        __syncthreads();   // all reads of this stage done before async overwrites it
    }

    // epilogue: D layout: vgpr r -> M=r (lanes 0-15) / M=8+r (lanes 16-31), N=lane%16
    #pragma unroll
    for (int i = 0; i < 2; ++i) {
        #pragma unroll
        for (int j = 0; j < 2; ++j) {
            int m0 = blockM + wm * 32 + i * 16 + ((lane >= 16) ? 8 : 0);
            int n  = blockN + wn * 32 + j * 16 + (lane & 15);
            float bv = bias ? bias[n] : 0.0f;
            #pragma unroll
            for (int r = 0; r < 8; ++r) {
                int m = m0 + r;
                float v = acc[i][j][r] + bv;
                if (addend) v += addend[(size_t)m * ldadd + n];
                C[(size_t)m * ldc + n] = v;
            }
        }
    }
}

// ---------------------------------------------------------------------------
// Small helper kernels
// ---------------------------------------------------------------------------
__global__ void cvt_f32_bf16(const float* __restrict__ in,
                             unsigned short* __restrict__ out, int n) {
    int i = blockIdx.x * blockDim.x + threadIdx.x;
    if (i < n) out[i] = f32_to_bf16(in[i]);
}

__global__ void bias_sum(const float* a, const float* b, float* o, int n) {
    int i = blockIdx.x * blockDim.x + threadIdx.x;
    if (i < n) o[i] = a[i] + b[i];
}

// x[b,s,:] = (s==0) ? features[b,:] : embed[captions[b,s-1],:]   (as bf16)
__global__ void build_x(const float* __restrict__ features,
                        const int* __restrict__ captions,
                        const float* __restrict__ embed,
                        unsigned short* __restrict__ x, int total) {
    int idx = blockIdx.x * blockDim.x + threadIdx.x;
    if (idx >= total) return;
    int e  = idx & 1023;
    int bs = idx >> 10;
    int s  = bs & 63;
    int b  = bs >> 6;
    float v = (s == 0) ? features[b * 1024 + e]
                       : embed[(size_t)captions[b * 64 + (s - 1)] * 1024 + e];
    x[idx] = f32_to_bf16(v);
}

__global__ void init_state(float* c, unsigned short* h, int n) {
    int i = blockIdx.x * blockDim.x + threadIdx.x;
    if (i < n) { c[i] = 0.0f; h[i] = 0; }
}

// gates (64x4096, i|f|g|o) -> update c, h; combined[b,t,:] = h + features[b,:]
// (attention collapses: softmax over singleton axis => context == features)
__global__ void lstm_cell(const float* __restrict__ gates,
                          float* __restrict__ c,
                          const float* __restrict__ features,
                          unsigned short* __restrict__ h_bf16,
                          unsigned short* __restrict__ combined, int t) {
    int idx = blockIdx.x * blockDim.x + threadIdx.x;   // 64*1024
    int b = idx >> 10, j = idx & 1023;
    const float* g = gates + (size_t)b * 4096;
    float i_ = g[j], f_ = g[1024 + j], gg = g[2048 + j], o_ = g[3072 + j];
    float si = 1.f / (1.f + __expf(-i_));
    float sf = 1.f / (1.f + __expf(-f_));
    float so = 1.f / (1.f + __expf(-o_));
    float cn = sf * c[idx] + si * tanhf(gg);
    c[idx] = cn;
    float h = so * tanhf(cn);
    h_bf16[idx] = f32_to_bf16(h);
    combined[((size_t)b * 64 + t) * 1024 + j] = f32_to_bf16(h + features[b * 1024 + j]);
}

// ---------------------------------------------------------------------------
// Launch
// ---------------------------------------------------------------------------
extern "C" void kernel_launch(void* const* d_in, const int* in_sizes, int n_in,
                              void* d_out, int out_size, void* d_ws, size_t ws_size,
                              hipStream_t stream) {
    (void)in_sizes; (void)n_in; (void)out_size; (void)ws_size;
    const float* features = (const float*)d_in[0];
    const int*   captions = (const int*)  d_in[1];
    const float* embed    = (const float*)d_in[2];
    const float* W_ih     = (const float*)d_in[3];
    const float* W_hh     = (const float*)d_in[4];
    const float* b_ih     = (const float*)d_in[5];
    const float* b_hh     = (const float*)d_in[6];
    const float* out_W    = (const float*)d_in[11];
    const float* out_b    = (const float*)d_in[12];
    float* out = (float*)d_out;

    // workspace carve-up (256B aligned)
    size_t off = 0;
    auto alloc = [&](size_t bytes) {
        off = (off + 255) & ~(size_t)255;
        void* p = (char*)d_ws + off;
        off += bytes;
        return p;
    };
    unsigned short* wih_bf  = (unsigned short*)alloc(4096 * 1024 * 2);
    unsigned short* whh_bf  = (unsigned short*)alloc(4096 * 1024 * 2);
    unsigned short* outw_bf = (unsigned short*)alloc((size_t)32000 * 1024 * 2);
    unsigned short* x_bf    = (unsigned short*)alloc((size_t)4096 * 1024 * 2);
    unsigned short* comb_bf = (unsigned short*)alloc((size_t)4096 * 1024 * 2);
    unsigned short* h_bf    = (unsigned short*)alloc(64 * 1024 * 2);
    float* xg    = (float*)alloc((size_t)4096 * 4096 * 4);
    float* gates = (float*)alloc(64 * 4096 * 4);
    float* cst   = (float*)alloc(64 * 1024 * 4);
    float* bsum  = (float*)alloc(4096 * 4);

    // 1) bf16 conversions of weights
    cvt_f32_bf16<<<(4096 * 1024 + 255) / 256, 256, 0, stream>>>(W_ih, wih_bf, 4096 * 1024);
    cvt_f32_bf16<<<(4096 * 1024 + 255) / 256, 256, 0, stream>>>(W_hh, whh_bf, 4096 * 1024);
    cvt_f32_bf16<<<(32000 * 1024 + 255) / 256, 256, 0, stream>>>(out_W, outw_bf, 32000 * 1024);
    bias_sum<<<16, 256, 0, stream>>>(b_ih, b_hh, bsum, 4096);

    // 2) embedding gather + input assembly
    build_x<<<(4096 * 1024 + 255) / 256, 256, 0, stream>>>(features, captions, embed,
                                                           x_bf, 4096 * 1024);

    // 3) xg = x @ W_ih^T + (b_ih + b_hh)   [M=4096,N=4096,K=1024]
    wmma_gemm_bf16<<<dim3(4096 / BN, 4096 / BM), 256, 0, stream>>>(
        x_bf, 1024, wih_bf, 1024, xg, 4096, bsum, nullptr, 0, 1024);

    // 4) LSTM scan (64 sequential steps)
    init_state<<<256, 256, 0, stream>>>(cst, h_bf, 64 * 1024);
    for (int t = 0; t < 64; ++t) {
        // gates = h @ W_hh^T + xg[:,t,:]   [M=64,N=4096,K=1024]
        wmma_gemm_bf16<<<dim3(4096 / BN, 64 / BM), 256, 0, stream>>>(
            h_bf, 1024, whh_bf, 1024, gates, 4096,
            nullptr, xg + (size_t)t * 4096, 64 * 4096, 1024);
        lstm_cell<<<256, 256, 0, stream>>>(gates, cst, features, h_bf, comb_bf, t);
    }

    // 5) out = combined @ out_W^T + out_b   [M=4096,N=32000,K=1024]
    wmma_gemm_bf16<<<dim3(32000 / BN, 4096 / BM), 256, 0, stream>>>(
        comb_bf, 1024, outw_bf, 1024, out, 32000, out_b, nullptr, 0, 1024);
}
